// MultiHeadCoAttention_88158498717883
// MI455X (gfx1250) — matrix operand order repaired and checked by reading, not compiled, not run on hardware
//
#include <hip/hip_runtime.h>
#include <hip/hip_bf16.h>
#include <math.h>

// MI455X (gfx1250) multi-head co-attention, flash-style, WMMA f16 + async-LDS staging.
// Pipeline:
//   1) cvt f32->f16 of activations + weights (vectorized)
//   2) WMMA GEMM projections (A tile staged in LDS via async-to-LDS; B fragments
//      double-buffered with buffer alternation -> no register copies)
//   3) flash co-attention twice; K/V chunks double-buffered in LDS via async-to-LDS,
//      shared by all 8 waves of the block (same bh, 128 query rows)
//   4) WMMA GEMM output projections -> f32 d_out (concatenated q_out | c_out)
// Workspace use: ~76 MB of d_ws.

#define DEV __device__ __forceinline__

typedef __attribute__((ext_vector_type(16))) _Float16 v16h;
typedef __attribute__((ext_vector_type(8)))  _Float16 h8;
typedef __attribute__((ext_vector_type(4)))  _Float16 h4;
typedef __attribute__((ext_vector_type(8)))  float    v8f;
typedef __attribute__((ext_vector_type(4)))  float    f4;
typedef __attribute__((ext_vector_type(4)))  int      v4i;

static constexpr int Bsz  = 2;
static constexpr int Lseq = 2048;
static constexpr int Dmod = 1024;
static constexpr int Hh   = 16;
static constexpr int DKh  = 64;
static constexpr int BH   = Bsz * Hh;                 // 32
static constexpr int BLD  = Bsz * Lseq * Dmod;        // 4194304
static constexpr int DD   = Dmod * Dmod;              // 1048576

DEV v8f wmma_f16(v16h a, v16h b, v8f c) {
  return __builtin_amdgcn_wmma_f32_16x16x32_f16(false, a, false, b, (short)0, c,
                                                false, false);
}

// ---- async global -> LDS copy (16 bytes per lane), with fallback ----------
DEV void async_cp16(const _Float16* g, _Float16* l) {
#if __has_builtin(__builtin_amdgcn_global_load_async_to_lds_b128)
  __builtin_amdgcn_global_load_async_to_lds_b128(
      (__attribute__((address_space(1))) v4i*)g,
      (__attribute__((address_space(3))) v4i*)l, 0, 0);
#else
  *(h8*)l = *(const h8*)g;
#endif
}

DEV void wait_async0() {
#if __has_builtin(__builtin_amdgcn_s_wait_asynccnt)
  __builtin_amdgcn_s_wait_asynccnt(0);
#else
  asm volatile("s_wait_asynccnt 0x0" ::: "memory");
#endif
}

DEV void wait_async2() {
#if __has_builtin(__builtin_amdgcn_s_wait_asynccnt)
  __builtin_amdgcn_s_wait_asynccnt(2);
#else
  asm volatile("s_wait_asynccnt 0x2" ::: "memory");
#endif
}

// A fragment, 16x32 f16 (MxK), row-major source with `stride` elements per row.
// ISA layout: lanes 0-15 hold M=lane, K={0..7,16..23}; lanes 16-31 K={8..15,24..31}.
DEV v16h load_frag_A(const _Float16* __restrict__ base, int stride, int lane) {
  const int m = lane & 15, half = lane >> 4;
  const _Float16* p = base + (size_t)m * stride + half * 8;
  h8 lo = *(const h8*)(p);        // K = half*8 + 0..7
  h8 hi = *(const h8*)(p + 16);   // K = half*8 + 16..23
  v16h a;
#pragma unroll
  for (int i = 0; i < 8; ++i) { a[i] = lo[i]; a[8 + i] = hi[i]; }
  return a;
}

// B fragment, 32x16 f16 (KxN) where B[k][n] = src[n][k] (src row-major, rows = n).
// ISA layout: lanes 0-15 hold N=lane, K={0..7,16..23}; lanes 16-31 K={8..15,24..31}.
DEV v16h load_frag_B(const _Float16* __restrict__ base, int stride, int lane) {
  const int n = lane & 15, half = lane >> 4;
  const _Float16* p = base + (size_t)n * stride + half * 8;
  h8 lo = *(const h8*)(p);
  h8 hi = *(const h8*)(p + 16);
  v16h b;
#pragma unroll
  for (int i = 0; i < 8; ++i) { b[i] = lo[i]; b[8 + i] = hi[i]; }
  return b;
}

__global__ void coattn_cvt_f16_kernel(const float* __restrict__ src,
                                      _Float16* __restrict__ dst, int n) {
  int i = (blockIdx.x * blockDim.x + threadIdx.x) * 4;
  const int stride = gridDim.x * blockDim.x * 4;
  for (; i < n; i += stride) {
    f4 v = *(const f4*)(src + i);
    h4 o;
#pragma unroll
    for (int k = 0; k < 4; ++k) o[k] = (_Float16)v[k];
    *(h4*)(dst + i) = o;
  }
}

// Y = X @ W^T + bias.  X: [M x K] f16 row-major, W: [N x K] f16 row-major.
// All 8 waves of a block share one 16-row A tile -> staged once in LDS (async).
// K-loop unrolled x2 with alternating B fragment buffers (no register copies).
// NOTE: deliberately overreads W by <=64 f16 on the final half-step; W buffers
// live in the middle of the d_ws arena so the overread stays in-bounds, and the
// fetched fragments are never consumed.
// mode 0: f16 heads [BH][Lseq][DKh];  1: f16 headsT [BH][DKh][Lseq];  2: f32 [MxN]
__global__ __launch_bounds__(256)
void coattn_gemm_wmma_kernel(const _Float16* __restrict__ X,
                             const _Float16* __restrict__ W,
                             const float* __restrict__ bias,
                             void* __restrict__ out,
                             int M, int K, int N, int mode) {
  __shared__ __align__(16) _Float16 Xs[16 * Dmod];   // 32 KB
  const int lane = threadIdx.x & 31;
  const int wid  = threadIdx.x >> 5;
  const int ntiles = N >> 6;
  const int tile = blockIdx.x * 8 + wid;
  const int mt = tile / ntiles;        // uniform across the block's 8 waves
  const int nt = tile % ntiles;
  if (mt * 16 >= M) return;            // block-uniform guard

  // stage the 16xK A tile (contiguous rows) into LDS, 16 B per thread per step
  {
    const _Float16* Xrow = X + (size_t)mt * 16 * K;
    const int chunks = (16 * K) / 8;                 // 16B chunks
#pragma unroll
    for (int i = 0; i < 8; ++i) {
      int c = (threadIdx.x + 256 * i);
      if (c < chunks) async_cp16(Xrow + (size_t)c * 8, Xs + (size_t)c * 8);
    }
    wait_async0();
    __syncthreads();
  }

  const _Float16* Wb = W + (size_t)nt * 64 * K;

  v8f zero = {};
  v8f acc[4];
#pragma unroll
  for (int t = 0; t < 4; ++t) acc[t] = zero;

  // alternating double-buffered B fragments; A fragments come from LDS
  v16h b0[4], b1[4];
#pragma unroll
  for (int t = 0; t < 4; ++t) b0[t] = load_frag_B(Wb + (size_t)t * 16 * K, K, lane);

  for (int k0 = 0; k0 < K; k0 += 64) {
    __builtin_prefetch(Wb + k0 + 256, 0, 3);
#pragma unroll
    for (int t = 0; t < 4; ++t)
      b1[t] = load_frag_B(Wb + (size_t)t * 16 * K + k0 + 32, K, lane);
    v16h a = load_frag_A(Xs + k0, K, lane);
#pragma unroll
    for (int t = 0; t < 4; ++t) acc[t] = wmma_f16(a, b0[t], acc[t]);

#pragma unroll
    for (int t = 0; t < 4; ++t)   // last-iter overread: harmless, unused
      b0[t] = load_frag_B(Wb + (size_t)t * 16 * K + k0 + 64, K, lane);
    a = load_frag_A(Xs + k0 + 32, K, lane);
#pragma unroll
    for (int t = 0; t < 4; ++t) acc[t] = wmma_f16(a, b1[t], acc[t]);
  }

  const int half = lane >> 4, nl = lane & 15;
#pragma unroll
  for (int t = 0; t < 4; ++t) {
    const int ng = nt * 64 + t * 16 + nl;
    const float bv = bias ? bias[ng] : 0.0f;
#pragma unroll
    for (int r = 0; r < 8; ++r) {
      const int m = mt * 16 + r + 8 * half;   // C/D layout: row = r + 8*(lane>=16)
      const float v = acc[t][r] + bv;
      if (mode == 2) {
        ((float*)out)[(size_t)m * N + ng] = v;
      } else {
        const int b = m / Lseq, l = m % Lseq;
        const int h = ng / DKh, dk = ng % DKh;
        const size_t idx = (mode == 0)
            ? ((size_t)(b * Hh + h) * Lseq + l) * DKh + dk
            : ((size_t)(b * Hh + h) * DKh + dk) * Lseq + l;
        ((_Float16*)out)[idx] = (_Float16)v;
      }
    }
  }
}

// Flash co-attention. Block = 8 waves, all on the same (b,h), 128 query rows.
// K/V chunks (32 context cols) are double-buffered in LDS via async-to-LDS and
// shared by all waves. Qh,Kh: [BH][L][64] f16; VT: [BH][64][L] f16.
// Out: f16 merged heads [B][L][D].
__global__ __launch_bounds__(256)
void coattn_flash_wmma_kernel(const _Float16* __restrict__ Qh,
                              const _Float16* __restrict__ Kh,
                              const _Float16* __restrict__ VT,
                              _Float16* __restrict__ Out) {
  __shared__ __align__(16) _Float16 Ks[2][32 * 64];   // 8 KB  (ctx row, d)
  __shared__ __align__(16) _Float16 Vs[2][64 * 32];   // 8 KB  (d, ctx col)
  __shared__ __align__(16) _Float16 Pbuf[8][16 * 32]; // 8 KB  per-wave P tiles

  const int tid  = threadIdx.x;
  const int lane = tid & 31;
  const int wid  = tid >> 5;
  const int bh = blockIdx.x >> 4;          // 16 blocks per (b,h)
  const int qb = blockIdx.x & 15;
  const int qt = qb * 8 + wid;             // this wave's 16-row query tile
  const int b = bh >> 4, h = bh & 15;
  const int half = lane >> 4, nl = lane & 15;

  const _Float16* Qb = Qh + (size_t)bh * Lseq * DKh + (size_t)qt * 16 * DKh;
  const _Float16* Kb = Kh + (size_t)bh * Lseq * DKh;
  const _Float16* Vb = VT + (size_t)bh * DKh * Lseq;

  // cooperative async stage of one 32-column chunk (4 KB K + 4 KB V)
  auto stage = [&](int buf, int j0) {
    {
      const int r = tid >> 3, c = (tid & 7) * 8;                 // K: 32x64
      async_cp16(Kb + (size_t)(j0 + r) * DKh + c, &Ks[buf][r * 64 + c]);
    }
    {
      const int d = tid >> 2, c = (tid & 3) * 8;                 // V: 64x32
      async_cp16(Vb + (size_t)d * Lseq + j0 + c, &Vs[buf][d * 32 + c]);
    }
  };

  stage(0, 0);

  const v16h aq0 = load_frag_A(Qb + 0,  DKh, lane);
  const v16h aq1 = load_frag_A(Qb + 32, DKh, lane);

  v8f zero = {};
  v8f o[4];
#pragma unroll
  for (int t = 0; t < 4; ++t) o[t] = zero;
  float mrow[8], lrow[8];
#pragma unroll
  for (int r = 0; r < 8; ++r) { mrow[r] = -__builtin_inff(); lrow[r] = 0.0f; }

  _Float16* myP = &Pbuf[wid][0];
  const float sc = 0.125f;                 // 1/sqrt(64)
  const int nchunk = Lseq / 32;            // 64

  for (int j = 0; j < nchunk; ++j) {
    const int buf = j & 1;
    if (j + 1 < nchunk) { stage(buf ^ 1, (j + 1) * 32); wait_async2(); }
    else                { wait_async0(); }
    __syncthreads();                       // chunk j resident in LDS for all waves

    const _Float16* Kc = &Ks[buf][0];      // [32][64]
    const _Float16* Vc = &Vs[buf][0];      // [64][32]

    // S(16x32) = Q(16x64) @ K_chunk^T, two 16-col subtiles, two K-steps each
    v8f s0 = zero, s1 = zero;
    s0 = wmma_f16(aq0, load_frag_B(Kc + 0,            64, lane), s0);
    s0 = wmma_f16(aq1, load_frag_B(Kc + 32,           64, lane), s0);
    s1 = wmma_f16(aq0, load_frag_B(Kc + 16 * 64,      64, lane), s1);
    s1 = wmma_f16(aq1, load_frag_B(Kc + 16 * 64 + 32, 64, lane), s1);

    // online softmax over the 32 chunk columns; stats per C-layout half-row
#pragma unroll
    for (int r = 0; r < 8; ++r) {
      float v0 = s0[r] * sc, v1 = s1[r] * sc;
      float mx = fmaxf(v0, v1);
#pragma unroll
      for (int d = 1; d < 16; d <<= 1) mx = fmaxf(mx, __shfl_xor(mx, d, 32));
      const float mnew  = fmaxf(mrow[r], mx);
      const float scale = __expf(mrow[r] - mnew);
      const float p0 = __expf(v0 - mnew), p1 = __expf(v1 - mnew);
      float ps = p0 + p1;
#pragma unroll
      for (int d = 1; d < 16; d <<= 1) ps += __shfl_xor(ps, d, 32);
      lrow[r] = lrow[r] * scale + ps;
      mrow[r] = mnew;
#pragma unroll
      for (int t = 0; t < 4; ++t) o[t][r] *= scale;
      const int row = r + 8 * half;
      myP[row * 32 + nl]      = (_Float16)p0;   // C-layout -> row-major LDS
      myP[row * 32 + 16 + nl] = (_Float16)p1;
    }
    asm volatile("s_wait_dscnt 0" ::: "memory");  // P visible before A-reload

    // o(16x64) += P(16x32) @ V_chunk(32x64)
    const v16h ap = load_frag_A(myP, 32, lane);
#pragma unroll
    for (int t = 0; t < 4; ++t)
      o[t] = wmma_f16(ap, load_frag_B(Vc + t * 16 * 32, 32, lane), o[t]);

    __syncthreads();                       // all waves done with buf before reuse
  }

#pragma unroll
  for (int r = 0; r < 8; ++r) {
    const float inv = 1.0f / lrow[r];
    const int lq = qt * 16 + r + 8 * half;
#pragma unroll
    for (int t = 0; t < 4; ++t) {
      const float v = o[t][r] * inv;
      Out[((size_t)b * Lseq + lq) * Dmod + h * DKh + t * 16 + nl] = (_Float16)v;
    }
  }
}

extern "C" void kernel_launch(void* const* d_in, const int* in_sizes, int n_in,
                              void* d_out, int out_size, void* d_ws, size_t ws_size,
                              hipStream_t stream) {
  (void)in_sizes; (void)n_in; (void)out_size; (void)ws_size;
  const float* query   = (const float*)d_in[0];
  const float* context = (const float*)d_in[1];
  const float* W[6]; const float* Bv[6];
  for (int i = 0; i < 6; ++i) {
    W[i]  = (const float*)d_in[2 + 2 * i];
    Bv[i] = (const float*)d_in[3 + 2 * i];
  }

  char* p = (char*)d_ws;
  auto take = [&](size_t elems) {
    _Float16* r = (_Float16*)p; p += elems * sizeof(_Float16); return r;
  };
  _Float16* q16  = take(BLD);
  _Float16* c16  = take(BLD);
  _Float16* w16[6];
  for (int i = 0; i < 6; ++i) w16[i] = take(DD);
  const size_t HLD = (size_t)BH * Lseq * DKh;     // == BLD elements
  _Float16* qh   = take(HLD);
  _Float16* ch   = take(HLD);
  _Float16* qvT  = take(HLD);
  _Float16* cvT  = take(HLD);
  _Float16* qatt = take(BLD);
  _Float16* catt = take(BLD);

  // 1) f32 -> f16 staging
  coattn_cvt_f16_kernel<<<1024, 256, 0, stream>>>(query,   q16, BLD);
  coattn_cvt_f16_kernel<<<1024, 256, 0, stream>>>(context, c16, BLD);
  for (int i = 0; i < 6; ++i)
    coattn_cvt_f16_kernel<<<512, 256, 0, stream>>>(W[i], w16[i], DD);

  // 2) projections
  const int M = Bsz * Lseq;                       // 4096
  const int gblocks = (M / 16) * (Dmod / 64) / 8; // 512 blocks x 8 waves
  coattn_gemm_wmma_kernel<<<gblocks, 256, 0, stream>>>(q16, w16[0], Bv[0], qh,  M, Dmod, Dmod, 0);
  coattn_gemm_wmma_kernel<<<gblocks, 256, 0, stream>>>(c16, w16[1], Bv[1], ch,  M, Dmod, Dmod, 0);
  coattn_gemm_wmma_kernel<<<gblocks, 256, 0, stream>>>(q16, w16[2], Bv[2], qvT, M, Dmod, Dmod, 1);
  coattn_gemm_wmma_kernel<<<gblocks, 256, 0, stream>>>(c16, w16[3], Bv[3], cvT, M, Dmod, Dmod, 1);

  // 3) flash co-attention: row-softmax branch, then column-softmax branch
  //    (column softmax of scores == flash attention with Q/K roles swapped)
  const int fblocks = BH * (Lseq / 128);          // 512
  coattn_flash_wmma_kernel<<<fblocks, 256, 0, stream>>>(qh, ch, cvT, qatt);
  coattn_flash_wmma_kernel<<<fblocks, 256, 0, stream>>>(ch, qh, qvT, catt);

  // 4) output projections, f32 results straight into d_out (q_out | c_out)
  float* out = (float*)d_out;
  coattn_gemm_wmma_kernel<<<gblocks, 256, 0, stream>>>(qatt, w16[4], Bv[4], out,       M, Dmod, Dmod, 2);
  coattn_gemm_wmma_kernel<<<gblocks, 256, 0, stream>>>(catt, w16[5], Bv[5], out + BLD, M, Dmod, Dmod, 2);
}
